// BERTSelfAttention_3994319585865
// MI455X (gfx1250) — matrix operand order repaired and checked by reading, not compiled, not run on hardware
//
#include <hip/hip_runtime.h>

#define HIDDEN   768
#define HEADS    12
#define HEAD_DIM 64
#define BATCH    2
#define SEQ      2048
#define ROWS     (BATCH * SEQ)   // 4096

typedef __attribute__((ext_vector_type(16))) __bf16        v16bf;
typedef __attribute__((ext_vector_type(8)))  float         v8f;
typedef __attribute__((ext_vector_type(4)))  unsigned int  u32x4;

union Frag {
    v16bf          v;
    unsigned short u[16];
    u32x4          q[2];
};

union Pack8 {
    unsigned short s[8];
    u32x4          q;
};

__device__ __forceinline__ unsigned short f2bf(float f) {
    unsigned int x = __float_as_uint(f);
    x += 0x7FFFu + ((x >> 16) & 1u);          // round-to-nearest-even
    return (unsigned short)(x >> 16);
}

__device__ __forceinline__ v8f vzero() {
    v8f z;
#pragma unroll
    for (int i = 0; i < 8; ++i) z[i] = 0.0f;
    return z;
}

__device__ __forceinline__ v8f wmma_bf16(const Frag& a, const Frag& b, v8f c) {
    return __builtin_amdgcn_wmma_f32_16x16x32_bf16(
        /*neg_a=*/false, a.v, /*neg_b=*/false, b.v,
        /*c_mod=*/(short)0, c, /*reuse_a=*/false, /*reuse_b=*/false);
}

// ---------------------------------------------------------------------------
// Kernel 0a: X (fp32) -> Xb (bf16), elementwise. 8 elements per thread.
// ---------------------------------------------------------------------------
__global__ __launch_bounds__(256) void cvt_x_kernel(
    const float* __restrict__ X, unsigned short* __restrict__ Xb)
{
    const size_t i = ((size_t)blockIdx.x * 256 + threadIdx.x) * 8;
    const float* p = X + i;
    Pack8 o;
#pragma unroll
    for (int j = 0; j < 8; ++j) o.s[j] = f2bf(p[j]);
    *(u32x4*)(Xb + i) = o.q;
}

// ---------------------------------------------------------------------------
// Kernel 0b: Wq/Wk/Wv (fp32, [K][N]) -> Wt (bf16, [mat][N][K]) transpose+convert.
// Each thread: 8 K-elements of one column; reads coalesced across lanes (n).
// ---------------------------------------------------------------------------
__global__ __launch_bounds__(256) void cvt_w_kernel(
    const float* __restrict__ Wq, const float* __restrict__ Wk,
    const float* __restrict__ Wv, unsigned short* __restrict__ Wt)
{
    const int tid = blockIdx.x * 256 + threadIdx.x;
    const int n   = tid % HIDDEN;            // column (fastest -> coalesced reads)
    const int t   = tid / HIDDEN;
    const int kb  = t % (HIDDEN / 8);        // 0..95
    const int mat = t / (HIDDEN / 8);        // 0..2
    const float* W = (mat == 0) ? Wq : (mat == 1) ? Wk : Wv;
    const float* p = W + (size_t)kb * 8 * HIDDEN + n;
    Pack8 o;
#pragma unroll
    for (int i = 0; i < 8; ++i) o.s[i] = f2bf(p[(size_t)i * HIDDEN]);
    *(u32x4*)(Wt + (size_t)mat * HIDDEN * HIDDEN + (size_t)n * HIDDEN + kb * 8) = o.q;
}

// ---------------------------------------------------------------------------
// Kernel 1: fused QKV projection, all-bf16 operands, bias added.
//   Q -> [B,H,S,D] row-major bf16
//   K -> [B,H,S,D] row-major bf16
//   V -> [B,H,D,S] (transposed) bf16
// One wave computes a 32x64 output tile (two 16-row sub-tiles sharing the four
// B fragments) with double-buffered fragments: while the 8 WMMAs of chunk k
// run, chunk k+1's 12 b128 loads are already in flight.
// 128 row-tiles x 12 col-tiles x 3 mats = 4608 waves.
// ---------------------------------------------------------------------------
__global__ __launch_bounds__(128) void qkv_proj_kernel(
    const unsigned short* __restrict__ Xb,   // [4096][768] bf16
    const unsigned short* __restrict__ Wt,   // [3][768 n][768 k] bf16
    const float* __restrict__ bq, const float* __restrict__ bk,
    const float* __restrict__ bv,
    unsigned short* __restrict__ qb,
    unsigned short* __restrict__ kbuf,
    unsigned short* __restrict__ vtb)
{
    const int lane = threadIdx.x & 31;
    const int widx = blockIdx.x * 4 + (threadIdx.x >> 5);
    const int mat  = widx / (128 * 12);
    const int rem  = widx - mat * (128 * 12);
    const int rowTile = rem / 12;            // 0..127 (32 rows each)
    const int colTile = rem - rowTile * 12;

    const unsigned short* W = Wt + (size_t)mat * HIDDEN * HIDDEN;
    const float* bias = (mat == 0) ? bq : (mat == 1) ? bk : bv;

    const int row32 = rowTile << 5;
    const int col64 = colTile << 6;
    const int l15   = lane & 15;
    const int kbA   = (lane & 16) ? 16 : 0;   // A-fragment K sub-base
    const int kbB   = (lane & 16) ? 16 : 0;   // B-fragment K sub-base

    v8f acc[8];
#pragma unroll
    for (int j = 0; j < 8; ++j) acc[j] = vzero();

    const unsigned short* xr0 = Xb + (size_t)(row32 + l15)      * HIDDEN + kbA;
    const unsigned short* xr1 = Xb + (size_t)(row32 + 16 + l15) * HIDDEN + kbA;
    const unsigned short* wb  = W  + (size_t)(col64 + l15)      * HIDDEN + kbB;

    Frag a0[2], a1[2], bf[2][4];
    // preload k-chunk 0 into slot 0
    a0[0].q[0] = *(const u32x4*)(xr0);
    a0[0].q[1] = *(const u32x4*)(xr0 + 8);
    a1[0].q[0] = *(const u32x4*)(xr1);
    a1[0].q[1] = *(const u32x4*)(xr1 + 8);
#pragma unroll
    for (int j = 0; j < 4; ++j) {
        bf[0][j].q[0] = *(const u32x4*)(wb + (size_t)j * 16 * HIDDEN);
        bf[0][j].q[1] = *(const u32x4*)(wb + (size_t)j * 16 * HIDDEN + 8);
    }

#pragma unroll
    for (int kc = 0; kc < HIDDEN / 32; ++kc) {      // 24 chunks
        const int s  = kc & 1;
        const int nx = s ^ 1;
        if (kc + 1 < HIDDEN / 32) {                 // prefetch next chunk
            const int kn = (kc + 1) * 32;
            a0[nx].q[0] = *(const u32x4*)(xr0 + kn);
            a0[nx].q[1] = *(const u32x4*)(xr0 + kn + 8);
            a1[nx].q[0] = *(const u32x4*)(xr1 + kn);
            a1[nx].q[1] = *(const u32x4*)(xr1 + kn + 8);
#pragma unroll
            for (int j = 0; j < 4; ++j) {
                bf[nx][j].q[0] = *(const u32x4*)(wb + (size_t)j * 16 * HIDDEN + kn);
                bf[nx][j].q[1] = *(const u32x4*)(wb + (size_t)j * 16 * HIDDEN + kn + 8);
            }
        }
#pragma unroll
        for (int j = 0; j < 4; ++j) {
            acc[j]     = wmma_bf16(a0[s], bf[s][j], acc[j]);
            acc[4 + j] = wmma_bf16(a1[s], bf[s][j], acc[4 + j]);
        }
    }

    // epilogue: bias, convert to bf16, scatter into attention layouts
#pragma unroll
    for (int half = 0; half < 2; ++half) {
#pragma unroll
        for (int j = 0; j < 4; ++j) {
            const int   n    = col64 + j * 16 + l15;
            const float bval = bias[n];
            const int   h    = n >> 6;
            const int   d    = n & 63;
#pragma unroll
            for (int r = 0; r < 8; ++r) {
                const int m  = row32 + half * 16 + r + ((lane & 16) ? 8 : 0);
                const int bb = m >> 11;          // batch
                const int sq = m & (SEQ - 1);    // sequence pos
                const unsigned short val = f2bf(acc[half * 4 + j][r] + bval);
                if (mat == 2) {
                    vtb[((size_t)(bb * HEADS + h) * HEAD_DIM + d) * SEQ + sq] = val;
                } else {
                    unsigned short* dst = (mat == 0) ? qb : kbuf;
                    dst[((size_t)(bb * HEADS + h) * SEQ + sq) * HEAD_DIM + d] = val;
                }
            }
        }
    }
}

// ---------------------------------------------------------------------------
// Kernel 2: flash attention. One wave owns 16 query rows of one (b,h) pair.
// Streams keys/values in 64-wide tiles; online softmax; all GEMMs via WMMA.
// LDS used only to transpose probabilities C-fragment -> A-fragment layout.
// ---------------------------------------------------------------------------
__global__ __launch_bounds__(128) void attn_kernel(
    const unsigned short* __restrict__ qb,
    const unsigned short* __restrict__ kbuf,
    const unsigned short* __restrict__ vtb,
    const float* __restrict__ mask,
    float* __restrict__ out)
{
    __shared__ __align__(16) unsigned short ldsP[4][16 * 80]; // ld=80 halfs/row

    const int lane = threadIdx.x & 31;
    const int wid  = threadIdx.x >> 5;
    const int widx = blockIdx.x * 4 + wid;
    const int rowTile = widx & 127;          // 128 row tiles per (b,h)
    const int bh      = widx >> 7;           // 0..23
    const int b       = bh / HEADS;
    const int h       = bh - b * HEADS;

    const unsigned short* Qh = qb   + (size_t)bh * SEQ * HEAD_DIM;
    const unsigned short* Kh = kbuf + (size_t)bh * SEQ * HEAD_DIM;
    const unsigned short* Vt = vtb  + (size_t)bh * HEAD_DIM * SEQ;
    const float*          mk = mask + (size_t)b * SEQ;
    unsigned short*      myP = ldsP[wid];

    const int hi  = (lane & 16) ? 1 : 0;
    const int l15 = lane & 15;

    // preload the two Q A-fragments (D = 0..31 and 32..63); kept in VGPRs
    Frag qa[2];
    {
        const unsigned short* qp =
            Qh + (size_t)(rowTile * 16 + l15) * HEAD_DIM + hi * 16;
#pragma unroll
        for (int c = 0; c < 2; ++c) {
            qa[c].q[0] = *(const u32x4*)(qp + c * 32);       // K=kb..kb+7
            qa[c].q[1] = *(const u32x4*)(qp + c * 32 + 8);   // K=kb+8..kb+15
        }
    }

    float mstat[8], lstat[8];
#pragma unroll
    for (int r = 0; r < 8; ++r) { mstat[r] = -1e30f; lstat[r] = 0.0f; }
    v8f acc[4];
#pragma unroll
    for (int j = 0; j < 4; ++j) acc[j] = vzero();

    for (int t0 = 0; t0 < SEQ; t0 += 64) {
        // ---- scores: S = Q (16x64) @ K^T (64 x 64keys), 4 key sub-tiles ----
        v8f sc[4];
#pragma unroll
        for (int j = 0; j < 4; ++j) {
            v8f s = vzero();
            const unsigned short* kp =
                Kh + (size_t)(t0 + j * 16 + l15) * HEAD_DIM + hi * 16;
#pragma unroll
            for (int c = 0; c < 2; ++c) {
                Frag kf;                                   // contiguous 16 halfs
                kf.q[0] = *(const u32x4*)(kp + c * 32);
                kf.q[1] = *(const u32x4*)(kp + c * 32 + 8);
                s = wmma_bf16(qa[c], kf, s);
            }
            sc[j] = s;
        }
        // ---- scale + additive mask (column-indexed == lane-indexed) ----
#pragma unroll
        for (int j = 0; j < 4; ++j) {
            const float mv = mk[t0 + j * 16 + l15];
#pragma unroll
            for (int r = 0; r < 8; ++r) sc[j][r] = sc[j][r] * 0.125f + mv;
        }
        // ---- online softmax: per-register row stats, 16-lane xor reductions ----
#pragma unroll
        for (int r = 0; r < 8; ++r) {
            float t = fmaxf(fmaxf(sc[0][r], sc[1][r]), fmaxf(sc[2][r], sc[3][r]));
#pragma unroll
            for (int off = 1; off < 16; off <<= 1)
                t = fmaxf(t, __shfl_xor(t, off, 32));
            const float mnew  = fmaxf(mstat[r], t);
            const float alpha = __expf(mstat[r] - mnew);
            mstat[r] = mnew;
            lstat[r] *= alpha;
#pragma unroll
            for (int j = 0; j < 4; ++j) acc[j][r] *= alpha;
            float psum = 0.0f;
#pragma unroll
            for (int j = 0; j < 4; ++j) {
                const float p = __expf(sc[j][r] - mnew);
                sc[j][r] = p;
                psum += p;
            }
#pragma unroll
            for (int off = 1; off < 16; off <<= 1)
                psum += __shfl_xor(psum, off, 32);
            lstat[r] += psum;
        }
        // ---- transpose P through LDS: C-fragment -> A-fragment layout ----
        {
            const int rbase = hi * 8;
#pragma unroll
            for (int j = 0; j < 4; ++j)
#pragma unroll
                for (int r = 0; r < 8; ++r)
                    myP[(rbase + r) * 80 + j * 16 + l15] = f2bf(sc[j][r]);
        }
        // DS ops from the same wave complete in order; reads below see writes.
        // ---- context: acc += P (16x64keys) @ V (64keys x 64), 4 d sub-tiles ----
#pragma unroll
        for (int c = 0; c < 2; ++c) {
            Frag pa;
            const unsigned short* pp = myP + l15 * 80 + c * 32 + hi * 16;
            pa.q[0] = *(const u32x4*)(pp);
            pa.q[1] = *(const u32x4*)(pp + 8);
#pragma unroll
            for (int j = 0; j < 4; ++j) {
                Frag vf;                                   // V^T: contiguous keys
                const unsigned short* vp =
                    Vt + (size_t)(j * 16 + l15) * SEQ + t0 + c * 32 + hi * 16;
                vf.q[0] = *(const u32x4*)(vp);
                vf.q[1] = *(const u32x4*)(vp + 8);
                acc[j] = wmma_bf16(pa, vf, acc[j]);
            }
        }
    }

    // ---- normalize and store to [B, S, H*D] fp32 (coalesced 64B per row/reg) ----
#pragma unroll
    for (int j = 0; j < 4; ++j) {
#pragma unroll
        for (int r = 0; r < 8; ++r) {
            const float val = acc[j][r] / lstat[r];
            const int   m   = rowTile * 16 + r + hi * 8;   // 0..2047
            out[((size_t)(b * SEQ + m)) * HIDDEN + h * HEAD_DIM + j * 16 + l15] = val;
        }
    }
}

// ---------------------------------------------------------------------------
extern "C" void kernel_launch(void* const* d_in, const int* in_sizes, int n_in,
                              void* d_out, int out_size, void* d_ws, size_t ws_size,
                              hipStream_t stream) {
    const float* X    = (const float*)d_in[0];   // [2,2048,768]
    const float* mask = (const float*)d_in[1];   // [2,1,1,2048]
    const float* Wq   = (const float*)d_in[2];
    const float* bq   = (const float*)d_in[3];
    const float* Wk   = (const float*)d_in[4];
    const float* bk   = (const float*)d_in[5];
    const float* Wv   = (const float*)d_in[6];
    const float* bv   = (const float*)d_in[7];
    float* out = (float*)d_out;

    // workspace layout (all bf16 halves):
    //   Xb  [4096*768], Wt [3*768*768], Q [4096*768], K [4096*768], Vt [4096*768]
    unsigned short* xb  = (unsigned short*)d_ws;
    unsigned short* wt  = xb  + (size_t)ROWS * HIDDEN;
    unsigned short* qb  = wt  + (size_t)3 * HIDDEN * HIDDEN;
    unsigned short* kb  = qb  + (size_t)ROWS * HIDDEN;
    unsigned short* vtb = kb  + (size_t)ROWS * HIDDEN;

    // 0a: X -> bf16 (4096*768/8 threads)
    cvt_x_kernel<<<(ROWS * HIDDEN / 8) / 256, 256, 0, stream>>>(X, xb);
    // 0b: W -> bf16 transposed [mat][n][k] (3*768*96 threads)
    cvt_w_kernel<<<(3 * HIDDEN * (HIDDEN / 8)) / 256, 256, 0, stream>>>(Wq, Wk, Wv, wt);
    // 1: QKV projection: 128 row-tiles x 12 col-tiles x 3 mats = 4608 waves
    qkv_proj_kernel<<<1152, 128, 0, stream>>>(xb, wt, bq, bk, bv, qb, kb, vtb);
    // 2: attention: 24 (b,h) x 128 row-tiles = 3072 waves
    attn_kernel<<<768, 128, 0, stream>>>(qb, kb, vtb, mask, out);
}
